// Compressor_16252156248239
// MI455X (gfx1250) — compile-verified, matrix operand order checked by reading, not run on hardware
//
#include <hip/hip_runtime.h>
#include <hip/hip_bf16.h>
#include <math.h>

// ---------------- problem constants (from reference) ----------------
#define SEQ    16384
#define DIMK   4096
#define TWOD   1024      // 2*D
#define HEADD  512       // D
#define NWIN   4096      // SEQ / R
#define RWIN   4
#define ROPE_D 64
#define NTOT   2048      // combined output columns [kv | score]
#define KCH    32        // K chunk in bf16 elements
#define LDB    40        // padded LDS row stride (80B: 16B aligned, bank-conflict-free)
#define NCHUNK (DIMK / KCH)
#define BUFELEMS (128 * LDB)   // bf16 elements per LDS buffer (10240 B)

// CDNA5 async global->LDS copy path (ASYNCcnt). Flip to 0 for plain-load fallback.
#define ASYNC_COPY 1

typedef __attribute__((ext_vector_type(16))) __bf16 bf16x16;
typedef __attribute__((ext_vector_type(8)))  __bf16 bf16x8;
typedef __attribute__((ext_vector_type(8)))  float  f32x8;

union FragBF16 { bf16x16 v; bf16x8 h[2]; };

// copy 16 bf16 (32B) global -> LDS, as two async B128 transfers
__device__ __forceinline__ void copy16_to_lds(__bf16* lds_dst, const __bf16* g) {
#if ASYNC_COPY
    const unsigned l0 = (unsigned)(size_t)lds_dst;   // low 32 bits = LDS byte offset
    const unsigned long long g0 = (unsigned long long)(size_t)g;
    asm volatile("global_load_async_to_lds_b128 %0, %1, off"
                 :: "v"(l0), "v"(g0) : "memory");
    asm volatile("global_load_async_to_lds_b128 %0, %1, off"
                 :: "v"(l0 + 16u), "v"(g0 + 16ull) : "memory");
#else
    *(bf16x8*)lds_dst       = *(const bf16x8*)g;
    *(bf16x8*)(lds_dst + 8) = *(const bf16x8*)(g + 8);
#endif
}

__device__ __forceinline__ void wait_async0() {
#if ASYNC_COPY
    asm volatile("s_wait_asynccnt 0x0" ::: "memory");
#endif
}

// 8 WMMAs of one K-chunk: wave tile 32x64 from LDS read bases rdA/rdB
__device__ __forceinline__ void compute_chunk(const __bf16* rdA, const __bf16* rdB,
                                              f32x8 (&acc)[2][4]) {
    FragBF16 a[2], b[4];
#pragma unroll
    for (int i = 0; i < 2; ++i) {
        const __bf16* p = rdA + i * (16 * LDB);
        a[i].h[0] = *(const bf16x8*)p;
        a[i].h[1] = *(const bf16x8*)(p + 8);
    }
#pragma unroll
    for (int j = 0; j < 4; ++j) {
        const __bf16* p = rdB + j * (16 * LDB);
        b[j].h[0] = *(const bf16x8*)p;
        b[j].h[1] = *(const bf16x8*)(p + 8);
    }
#pragma unroll
    for (int i = 0; i < 2; ++i)
#pragma unroll
        for (int j = 0; j < 4; ++j)
            acc[i][j] = __builtin_amdgcn_wmma_f32_16x16x32_bf16(
                false, a[i].v, false, b[j].v,
                (short)0, acc[i][j], false, false);
}

// =====================================================================
// Stage 0: fp32 -> bf16 (8 elems/thread; all sizes divide evenly)
// =====================================================================
__global__ __launch_bounds__(256)
void cvt_f32_to_bf16_kernel(const float* __restrict__ src,
                            __bf16* __restrict__ dst) {
    const size_t i = ((size_t)blockIdx.x * 256 + threadIdx.x) * 8;
    float f[8] __attribute__((aligned(16)));
    *(float4*)&f[0] = *(const float4*)(src + i);
    *(float4*)&f[4] = *(const float4*)(src + i + 4);
    __bf16 h[8] __attribute__((aligned(16)));
#pragma unroll
    for (int e = 0; e < 8; ++e) h[e] = (__bf16)f[e];
    *(bf16x8*)(dst + i) = *(bf16x8*)&h[0];
}

// =====================================================================
// Stage 1: dual GEMM out[s,o] = sum_k x[s,k]*W[o,k], bf16 WMMA, f32 acc.
// Grid: blockIdx.x = N slab (16, fastest -> co-launched blocks share x rows,
// weights L2-resident), blockIdx.y = M slab (128).
// Block tile 128x128, 8 waves (4Mx2N), wave tile 32x64 = 2x4 WMMA accs.
// Double-buffered LDS fed by async global->LDS B128 copies; buffer toggle
// is a pointer +/- delta (single-version loop -> accumulators stay pinned).
// =====================================================================
__global__ __launch_bounds__(256)
void gemm_bf16_wmma_kernel(const __bf16* __restrict__ xb,
                           const __bf16* __restrict__ wb,
                           float* __restrict__ kv_out,
                           float* __restrict__ sc_out) {
    __shared__ __bf16 As[2 * BUFELEMS];
    __shared__ __bf16 Bs[2 * BUFELEMS];

    const int bn_global = blockIdx.x * 128;   // 0..2047 combined columns
    const int bm        = blockIdx.y * 128;

    const int tid  = threadIdx.x;
    const int lane = tid & 31;
    const int wv   = tid >> 5;     // 0..7
    const int wm   = wv >> 1;      // 0..3 : 32-row slice
    const int wn   = wv & 1;       // 0..1 : 64-col slice

    // copy mapping: thread t -> tile row t/2, 16-elem (32B) k-part (t&1)*16
    const int crow = tid >> 1;
    const int ckp  = (tid & 1) * 16;
    const __bf16* ga = xb + (size_t)(bm + crow) * DIMK + ckp;
    const __bf16* gb = wb + (size_t)(bn_global + crow) * DIMK + ckp;
    const int lofs = crow * LDB + ckp;

    f32x8 acc[2][4];
    const f32x8 zero = {0.f,0.f,0.f,0.f,0.f,0.f,0.f,0.f};
#pragma unroll
    for (int i = 0; i < 2; ++i)
#pragma unroll
        for (int j = 0; j < 4; ++j) acc[i][j] = zero;

    const int r0   = lane & 15;
    const int ksel = (lane >> 4) * 16;   // lanes 16..31 hold K=16..31

    // read bases (buffer 0) and write targets (buffer 1) as toggling pointers
    const __bf16* rdA = &As[(wm * 32 + r0) * LDB + ksel];
    const __bf16* rdB = &Bs[(wn * 64 + r0) * LDB + ksel];
    __bf16* wrA = &As[BUFELEMS + lofs];
    __bf16* wrB = &Bs[BUFELEMS + lofs];
    int delta = BUFELEMS;

    // prime the pipeline: chunk 0 -> buffer 0
    copy16_to_lds(&As[lofs], ga);
    copy16_to_lds(&Bs[lofs], gb);

#pragma unroll 1
    for (int ch = 0; ch < NCHUNK - 1; ++ch) {
        wait_async0();        // own async copies landed
        __syncthreads();      // => all waves' copies landed; prev compute done
        copy16_to_lds(wrA, ga + (size_t)(ch + 1) * KCH);
        copy16_to_lds(wrB, gb + (size_t)(ch + 1) * KCH);
        compute_chunk(rdA, rdB, acc);
        rdA += delta;  rdB += delta;
        wrA -= delta;  wrB -= delta;
        delta = -delta;
    }
    // last chunk: no further copy
    wait_async0();
    __syncthreads();
    compute_chunk(rdA, rdB, acc);

    // epilogue: C/D layout lane%16 = N, vgpr v -> M = v + 8*(lane>=16)
    const bool is_kv = (bn_global < TWOD);
    float* outp      = is_kv ? kv_out : sc_out;
    const int nb     = is_kv ? bn_global : (bn_global - TWOD);
    const int mbase  = bm + wm * 32 + ((lane >> 4) << 3);
    const int nbase  = nb + wn * 64 + (lane & 15);
#pragma unroll
    for (int i = 0; i < 2; ++i)
#pragma unroll
        for (int j = 0; j < 4; ++j)
#pragma unroll
            for (int v = 0; v < 8; ++v)
                outp[(size_t)(mbase + i * 16 + v) * TWOD + (nbase + j * 16)] =
                    acc[i][j][v];
}

// =====================================================================
// Stage 2: overlap transform + per-dim softmax pooling + RMSNorm + RoPE
// =====================================================================
__global__ __launch_bounds__(256)
void compress_pool_kernel(const float* __restrict__ kv,
                          const float* __restrict__ sc,
                          const float* __restrict__ ape,
                          const float* __restrict__ norm_w,
                          const float* __restrict__ fcos,
                          const float* __restrict__ fsin,
                          float* __restrict__ out) {
    const int n = blockIdx.x;
    const int t = threadIdx.x;
    __shared__ float comp_s[HEADD];
    __shared__ float red[256];

    float partial = 0.0f;
#pragma unroll
    for (int half = 0; half < 2; ++half) {
        const int d = t + half * 256;
        float sv[8], kvv[8];
#pragma unroll
        for (int w = 0; w < 4; ++w) {        // prev window, first-half dims
            if (n > 0) {
                const size_t row = (size_t)(n - 1) * RWIN + w;
                sv[w]  = sc[row * TWOD + d] + ape[w * TWOD + d];
                kvv[w] = kv[row * TWOD + d];
            } else {
                sv[w]  = -__builtin_inff();
                kvv[w] = 0.0f;
            }
        }
#pragma unroll
        for (int w = 0; w < 4; ++w) {        // current window, second-half dims
            const size_t row = (size_t)n * RWIN + w;
            sv[4 + w]  = sc[row * TWOD + HEADD + d] + ape[w * TWOD + HEADD + d];
            kvv[4 + w] = kv[row * TWOD + HEADD + d];
        }
        float m = sv[0];
#pragma unroll
        for (int w = 1; w < 8; ++w) m = fmaxf(m, sv[w]);
        float ssum = 0.0f, accv = 0.0f;
#pragma unroll
        for (int w = 0; w < 8; ++w) {
            const float p = __expf(sv[w] - m);
            ssum += p;
            accv += kvv[w] * p;
        }
        const float c = accv / ssum;
        comp_s[d] = c;
        partial += c * c;
    }

    red[t] = partial;
    __syncthreads();
#pragma unroll
    for (int off = 128; off > 0; off >>= 1) {
        if (t < off) red[t] += red[t + off];
        __syncthreads();
    }
    const float rms = rsqrtf(red[0] * (1.0f / (float)HEADD) + 1e-6f);

#pragma unroll
    for (int half = 0; half < 2; ++half) {
        const int d = t + half * 256;
        comp_s[d] = comp_s[d] * rms * norm_w[d];
    }
    __syncthreads();

#pragma unroll
    for (int half = 0; half < 2; ++half) {
        const int d = t + half * 256;
        float o;
        if (d < HEADD - ROPE_D) {
            o = comp_s[d];
        } else {
            const int i = (d - (HEADD - ROPE_D)) >> 1;
            const float cc = fcos[(size_t)(RWIN * n) * (ROPE_D / 2) + i];
            const float ss = fsin[(size_t)(RWIN * n) * (ROPE_D / 2) + i];
            if ((d & 1) == 0) o = comp_s[d] * cc - comp_s[d + 1] * ss;
            else              o = comp_s[d - 1] * ss + comp_s[d] * cc;
        }
        out[(size_t)n * HEADD + d] = o;
    }
}

extern "C" void kernel_launch(void* const* d_in, const int* in_sizes, int n_in,
                              void* d_out, int out_size, void* d_ws, size_t ws_size,
                              hipStream_t stream) {
    const float* x      = (const float*)d_in[0];
    const float* wkv_w  = (const float*)d_in[1];
    const float* wg_w   = (const float*)d_in[2];
    const float* ape    = (const float*)d_in[3];
    const float* norm_w = (const float*)d_in[4];
    const float* fcos   = (const float*)d_in[5];
    const float* fsin   = (const float*)d_in[6];
    // d_in[7] = compress_state, d_in[8] = start_pos : unused in prefill path
    float* out = (float*)d_out;

    // workspace layout (~272 MiB):
    //   xb: bf16 [SEQ x DIMK]  (128 MiB, fits L2 -> reused across N slabs)
    //   wb: bf16 [2048 x DIMK] (16 MiB, combined [kv;gate], L2-resident)
    //   kv_ws / sc_ws: fp32 [SEQ x 1024] each (64 MiB each)
    __bf16* xb   = (__bf16*)d_ws;
    __bf16* wb   = xb + (size_t)SEQ * DIMK;
    float* kv_ws = (float*)(wb + (size_t)NTOT * DIMK);
    float* sc_ws = kv_ws + (size_t)SEQ * TWOD;

    cvt_f32_to_bf16_kernel<<<(SEQ * (DIMK / 8)) / 256, 256, 0, stream>>>(x, xb);
    cvt_f32_to_bf16_kernel<<<(TWOD * (DIMK / 8)) / 256, 256, 0, stream>>>(wkv_w, wb);
    cvt_f32_to_bf16_kernel<<<(TWOD * (DIMK / 8)) / 256, 256, 0, stream>>>(
        wg_w, wb + (size_t)TWOD * DIMK);

    dim3 grid1(NTOT / 128, SEQ / 128);   // x = 16 N slabs (fastest), y = 128 M slabs
    gemm_bf16_wmma_kernel<<<grid1, 256, 0, stream>>>(xb, wb, kv_ws, sc_ws);

    compress_pool_kernel<<<NWIN, 256, 0, stream>>>(kv_ws, sc_ws, ape, norm_w,
                                                   fcos, fsin, out);
}
